// MultiHeadAttention_46067819217464
// MI455X (gfx1250) — matrix operand order repaired and checked by reading, not compiled, not run on hardware
//
#include <hip/hip_runtime.h>

typedef __attribute__((ext_vector_type(16))) _Float16 v16h;
typedef __attribute__((ext_vector_type(8)))  _Float16 v8h;
typedef __attribute__((ext_vector_type(4)))  _Float16 v4h;
typedef __attribute__((ext_vector_type(8)))  float    v8f;
typedef __attribute__((ext_vector_type(4)))  float    v4f;
typedef __attribute__((ext_vector_type(4)))  unsigned int v4u;
typedef __attribute__((ext_vector_type(4)))  int      v4i;
typedef __attribute__((ext_vector_type(8)))  int      v8i;

// Problem constants (from reference)
constexpr int kB    = 4;
constexpr int kS    = 2048;
constexpr int kH    = 16;
constexpr int kHD   = 64;
constexpr int kD    = 1024;     // d_model == d_hid
constexpr int kNQKV = 3072;
constexpr int kM    = kB * kS;  // 8192 rows

// GEMM tiling: 128x128 block tile, 8 waves (4x2), each wave 32x64 (2x4 WMMA frags)
constexpr int TM = 128, TN = 128, TK = 32, LDSK = 40;  // 40 halves = 80B pitch

// ---- CDNA5 async global->LDS staging (GLOBAL_LOAD_ASYNC_TO_LDS_B128) -------
#define USE_ASYNC_LDS 1

__device__ __forceinline__ void async_copy_b128(_Float16* lds_dst, const _Float16* gsrc) {
#if USE_ASYNC_LDS
  unsigned int laddr = (unsigned int)(size_t)lds_dst;  // flat->LDS offset truncation
  asm volatile("global_load_async_to_lds_b128 %0, %1, off"
               :
               : "v"(laddr), "v"(gsrc)
               : "memory");
#else
  *(v8h*)lds_dst = *(const v8h*)gsrc;
#endif
}

__device__ __forceinline__ void async_fence() {
#if USE_ASYNC_LDS
  asm volatile("s_wait_asynccnt 0x0" ::: "memory");
#endif
}

// ---- CDNA5 Tensor Data Mover staging (TENSOR_LOAD_TO_LDS, TENSORcnt) -------
#ifndef USE_TDM
#define USE_TDM 1
#endif
#if USE_TDM && defined(__has_builtin)
#if !__has_builtin(__builtin_amdgcn_tensor_load_to_lds) || \
    !__has_builtin(__builtin_amdgcn_s_wait_tensorcnt)
#undef USE_TDM
#define USE_TDM 0
#endif
#endif
#if defined(__has_include)
#if __has_include(<hip/amd_detail/amd_gfx1250_TDM.h>)
#define TDM_SIX_ARGS 1   // therock-10.0 headers -> 6-arg builtin
#else
#define TDM_SIX_ARGS 0   // ROCm 7.2 -> 5-arg builtin
#endif
#else
#define TDM_SIX_ARGS 0
#endif

#if USE_TDM
// 2D tile DMA, data_size = 8 bytes. Dims/strides in 8-byte units.
// D# bit packing per CDNA5 ISA ch.8 (group0: count/type/addresses; group1: dims).
__device__ __forceinline__ void tdm_load_2d(unsigned lds_off, const void* gaddr,
                                            unsigned tensor_d0, unsigned tensor_d1,
                                            unsigned tile_d0, unsigned tile_d1,
                                            unsigned stride0) {
  unsigned long long ga = (unsigned long long)gaddr;
  v4u g0;
  g0[0] = 1u;                                              // count=1 (valid), user mode
  g0[1] = lds_off;                                         // lds_addr
  g0[2] = (unsigned)(ga & 0xffffffffu);                    // global_addr[31:0]
  g0[3] = (unsigned)((ga >> 32) & 0x01ffffffu) | (2u << 30); // addr[56:32] | type=2
  v8i g1;
  g1[0] = (int)(3u << 16);                                 // wg_mask=0, data_size=8B
  g1[1] = (int)((tensor_d0 & 0xffffu) << 16);              // tensor_dim0[15:0]
  g1[2] = (int)(((tensor_d0 >> 16) & 0xffffu) | ((tensor_d1 & 0xffffu) << 16));
  g1[3] = (int)(((tensor_d1 >> 16) & 0xffffu) | ((tile_d0 & 0xffffu) << 16));
  g1[4] = (int)(tile_d1 & 0xffffu);                        // tile_dim1, tile_dim2=0
  g1[5] = (int)stride0;                                    // tensor_dim0_stride[31:0]
  g1[6] = 0;
  g1[7] = 0;
  v4i z4 = {0, 0, 0, 0};
#if TDM_SIX_ARGS
  v8i z8 = {0, 0, 0, 0, 0, 0, 0, 0};
  __builtin_amdgcn_tensor_load_to_lds(g0, g1, z4, z4, z8, 0);
#else
  __builtin_amdgcn_tensor_load_to_lds(g0, g1, z4, z4, 0);
#endif
}
#endif  // USE_TDM

__device__ __forceinline__ v16h cat16(v8h lo, v8h hi) {
  return __builtin_shufflevector(lo, hi, 0,1,2,3,4,5,6,7,8,9,10,11,12,13,14,15);
}

// A-fragment (16x32 f16): halves 0..15 = k[hi*16..hi*16+15] (contiguous)
__device__ __forceinline__ v16h ldfrag_a(const _Float16* p, int hi) {
  v8h lo = *(const v8h*)(p + hi * 16);
  v8h hh = *(const v8h*)(p + hi * 16 + 8);
  return cat16(lo, hh);
}

// B-fragment (32x16 f16): lane = column; 16 contiguous k-halves at p
__device__ __forceinline__ v16h ldfrag_b(const _Float16* p) {
  v8h lo = *(const v8h*)(p);
  v8h hh = *(const v8h*)(p + 8);
  return cat16(lo, hh);
}

__device__ __forceinline__ v8f wmma16x16x32(v16h a, v16h b, v8f c) {
  return __builtin_amdgcn_wmma_f32_16x16x32_f16(false, a, false, b, (short)0, c,
                                                false, false);
}

// ---------------------------------------------------------------------------
// fp32 -> f16 conversion, 4 elements/thread (n4 = n/4)
// ---------------------------------------------------------------------------
__global__ __launch_bounds__(256) void f32_to_f16_kernel(const float* __restrict__ src,
                                                         _Float16* __restrict__ dst,
                                                         int n4) {
  int i = blockIdx.x * blockDim.x + threadIdx.x;
  int stride = gridDim.x * blockDim.x;
  for (; i < n4; i += stride) {
    v4f s = ((const v4f*)src)[i];
    v4h d;
#pragma unroll
    for (int j = 0; j < 4; ++j) d[j] = (_Float16)s[j];
    ((v4h*)dst)[i] = d;
  }
}

// ---------------------------------------------------------------------------
// Shared GEMM mainloop: C(128x128) += A[m0:,k] * W[n0:,k]^T (both [rows][1024] f16)
// ---------------------------------------------------------------------------
__device__ __forceinline__ void gemm_tile_128x128(const _Float16* __restrict__ A,
                                                  const _Float16* __restrict__ W,
                                                  int m0, int n0,
                                                  _Float16* As, _Float16* Bs,
                                                  v8f acc[2][4]) {
  const int tid = threadIdx.x;
  const int wave = tid >> 5, lane = tid & 31;
  const int wm = wave >> 1, wn = wave & 1;
  const int ln = lane & 15, hi = lane >> 4;

  for (int k0 = 0; k0 < kD; k0 += TK) {
#pragma unroll
    for (int it = 0; it < 2; ++it) {
      int slot = tid + it * 256;
      int r = slot >> 2, c = (slot & 3) * 8;
      async_copy_b128(As + r * LDSK + c, A + (size_t)(m0 + r) * kD + k0 + c);
      async_copy_b128(Bs + r * LDSK + c, W + (size_t)(n0 + r) * kD + k0 + c);
    }
    if (k0 + TK < kD) {
      __builtin_prefetch(A + (size_t)(m0 + (tid >> 2)) * kD + k0 + TK, 0, 3);
      __builtin_prefetch(W + (size_t)(n0 + (tid >> 2)) * kD + k0 + TK, 0, 3);
    }
    async_fence();
    __syncthreads();

    v16h bf[4];
#pragma unroll
    for (int tn = 0; tn < 4; ++tn)
      bf[tn] = ldfrag_b(Bs + (wn * 64 + tn * 16 + ln) * LDSK + hi * 16);
#pragma unroll
    for (int tm = 0; tm < 2; ++tm) {
      v16h a = ldfrag_a(As + (wm * 32 + tm * 16 + ln) * LDSK, hi);
#pragma unroll
      for (int tn = 0; tn < 4; ++tn)
        acc[tm][tn] = wmma16x16x32(a, bf[tn], acc[tm][tn]);
    }
    __syncthreads();
  }
}

// ---------------------------------------------------------------------------
// Kernel 1: QKV projection. Layouts: Q,K regions [B][H][S][64] f16 (Q scaled 1/8),
// V region stored TRANSPOSED: [B][H][64][S] f16 so flash staging is contiguous.
// ---------------------------------------------------------------------------
__global__ __launch_bounds__(256) void qkv_gemm_kernel(const _Float16* __restrict__ xh,
                                                       const _Float16* __restrict__ wh,
                                                       _Float16* __restrict__ qkv) {
  __shared__ alignas(16) _Float16 As[TM * LDSK];
  __shared__ alignas(16) _Float16 Bs[TN * LDSK];
  const int m0 = blockIdx.x * TM, n0 = blockIdx.y * TN;
  v8f acc[2][4] = {};
  gemm_tile_128x128(xh, wh, m0, n0, As, Bs, acc);

  const int tid = threadIdx.x, wave = tid >> 5, lane = tid & 31;
  const int wm = wave >> 1, wn = wave & 1;
  const int n = lane & 15, hi = lane >> 4;
  const size_t region = (size_t)kB * kH * kS * kHD;
#pragma unroll
  for (int tm = 0; tm < 2; ++tm)
#pragma unroll
    for (int tn = 0; tn < 4; ++tn) {
      int e = n0 + wn * 64 + tn * 16 + n;          // column in [0, 3072)
      int h = e / 192, rem = e % 192;              // head / intra-head
      int which = rem / 64, d = rem % 64;          // 0=Q 1=K 2=V
      float sc = (which == 0) ? 0.125f : 1.0f;     // fold 1/sqrt(64) into Q
#pragma unroll
      for (int r = 0; r < 8; ++r) {
        int mrow = m0 + wm * 32 + tm * 16 + r + 8 * hi;
        int b = mrow >> 11, s = mrow & (kS - 1);
        size_t idx;
        if (which == 2)  // V^T: [b][h][d][s]
          idx = 2 * region + (((size_t)b * kH + h) * kHD + d) * kS + s;
        else             // Q/K: [b][h][s][d]
          idx = (((size_t)which * kB + b) * (size_t)kH + h) * ((size_t)kS * kHD) +
                (size_t)s * kHD + d;
        qkv[idx] = (_Float16)(acc[tm][tn][r] * sc);
      }
    }
}

// ---------------------------------------------------------------------------
// Kernel 2: flash attention.  Block = one (b,h) x 128 query rows; 8 waves,
// each wave owns 16 query rows and streams all 2048 keys in steps of 32.
// K tile (32x64) and V^T tile (64x32) staged by the Tensor Data Mover.
// ---------------------------------------------------------------------------
__global__ __launch_bounds__(256) void flash_attn_kernel(const _Float16* __restrict__ qkvb,
                                                         const int* __restrict__ mask,
                                                         _Float16* __restrict__ vals) {
  __shared__ alignas(16) _Float16 Ks[32 * 64];       // K tile  [kv][d]
  __shared__ alignas(16) _Float16 Vt[64 * 32];       // V^T tile [d][kv]
  __shared__ alignas(16) _Float16 Ps[8 * 16 * 32];   // per-wave P bounce buffer

  const int b = blockIdx.z, h = blockIdx.y;
  const int q0 = blockIdx.x * 128;
  const int tid = threadIdx.x, wave = tid >> 5, lane = tid & 31;
  const int n = lane & 15, hi = lane >> 4;
  const size_t region = (size_t)kB * kH * kS * kHD;

  const _Float16* Qb = qkvb + ((size_t)b * kH + h) * ((size_t)kS * kHD);
  const _Float16* Kb = qkvb + region + ((size_t)b * kH + h) * ((size_t)kS * kHD);
  const _Float16* Vg = qkvb + 2 * region + ((size_t)b * kH + h) * ((size_t)kHD * kS);

  // Q fragments straight from global (row = lane&15 of this wave's 16 rows)
  v16h qa[2];
  {
    const _Float16* qr = Qb + (size_t)(q0 + wave * 16 + n) * kHD;
#pragma unroll
    for (int c = 0; c < 2; ++c) qa[c] = ldfrag_a(qr + c * 32, hi);
  }

  float m_st[8], l_st[8];
  v8f accd[4] = {};
#pragma unroll
  for (int r = 0; r < 8; ++r) { m_st[r] = -3.0e38f; l_st[r] = 0.0f; }

  _Float16* myP = Ps + wave * (16 * 32);
  const int* mrow = mask + (size_t)b * kS;
#if USE_TDM
  const unsigned ks_off = (unsigned)(size_t)&Ks[0];
  const unsigned vt_off = (unsigned)(size_t)&Vt[0];
#endif

  for (int k0 = 0; k0 < kS; k0 += 32) {
#if USE_TDM
    if (tid < 32) {  // one wave issues both tile DMAs (TENSORcnt)
      // K: 32 rows x 128B, row stride 128B (=16 x 8B units)
      tdm_load_2d(ks_off, Kb + (size_t)k0 * kHD, 16, kS, 16, 32, 16);
      // V^T: 64 rows x 64B from [64][S], row stride 4096B (=512 units)
      tdm_load_2d(vt_off, Vg + k0, 512, kHD, 8, 64, 512);
      __builtin_amdgcn_s_wait_tensorcnt(0);
    }
#else
    {
      int kv = tid >> 3, c = (tid & 7) * 8;
      async_copy_b128(Ks + kv * 64 + c, Kb + (size_t)(k0 + kv) * kHD + c);
    }
    {
      int d = tid >> 2, c = (tid & 3) * 8;
      async_copy_b128(Vt + d * 32 + c, Vg + (size_t)d * kS + k0 + c);
    }
    async_fence();
#endif
    if (k0 + 32 < kS) {
      __builtin_prefetch(Kb + (size_t)(k0 + 32 + (tid >> 3)) * kHD, 0, 3);
      __builtin_prefetch(Vg + (size_t)(tid >> 2) * kS + k0 + 32, 0, 3);
    }
    __syncthreads();

    // S = Q * K^T for 32 keys: two 16-key output tiles, d accumulated over 2 chunks
    v8f sc[2] = {};
#pragma unroll
    for (int t = 0; t < 2; ++t)
#pragma unroll
      for (int c = 0; c < 2; ++c) {
        v16h bk = ldfrag_b(Ks + (t * 16 + n) * 64 + c * 32 + hi * 16);
        sc[t] = wmma16x16x32(qa[c], bk, sc[t]);
      }

    // mask bias (column-only)
    const float bias0 = mrow[k0 + n] ? 0.0f : -9.0e15f;
    const float bias1 = mrow[k0 + 16 + n] ? 0.0f : -9.0e15f;

    // online softmax; D-frag row m = r + 8*hi spans 16 lanes -> xor reductions
    float mloc[8];
#pragma unroll
    for (int r = 0; r < 8; ++r) {
      sc[0][r] += bias0;
      sc[1][r] += bias1;
      mloc[r] = fmaxf(sc[0][r], sc[1][r]);
    }
#pragma unroll
    for (int xm = 1; xm < 16; xm <<= 1)
#pragma unroll
      for (int r = 0; r < 8; ++r)
        mloc[r] = fmaxf(mloc[r], __shfl_xor(mloc[r], xm, 32));

    float rs[8], scale[8];
#pragma unroll
    for (int r = 0; r < 8; ++r) {
      float mn = fmaxf(m_st[r], mloc[r]);
      scale[r] = __expf(m_st[r] - mn);
      m_st[r] = mn;
      float p0 = __expf(sc[0][r] - mn);
      float p1 = __expf(sc[1][r] - mn);
      sc[0][r] = p0;
      sc[1][r] = p1;
      rs[r] = p0 + p1;
    }
#pragma unroll
    for (int xm = 1; xm < 16; xm <<= 1)
#pragma unroll
      for (int r = 0; r < 8; ++r) rs[r] += __shfl_xor(rs[r], xm, 32);
#pragma unroll
    for (int r = 0; r < 8; ++r) l_st[r] = l_st[r] * scale[r] + rs[r];
#pragma unroll
    for (int t = 0; t < 4; ++t)
#pragma unroll
      for (int r = 0; r < 8; ++r) accd[t][r] *= scale[r];

    // Re-layout P: f32 D-frag -> f16 A-frag via per-wave LDS (in-order per wave)
#pragma unroll
    for (int r = 0; r < 8; ++r) {
      myP[(r + 8 * hi) * 32 + n]      = (_Float16)sc[0][r];
      myP[(r + 8 * hi) * 32 + 16 + n] = (_Float16)sc[1][r];
    }
    v16h pa = ldfrag_a(myP + n * 32, hi);

    // acc += P(16x32) * V(32x64): 4 n-tiles of 16
#pragma unroll
    for (int t = 0; t < 4; ++t) {
      v16h bv = ldfrag_b(Vt + (t * 16 + n) * 32 + hi * 16);
      accd[t] = wmma16x16x32(pa, bv, accd[t]);
    }
    __syncthreads();
  }

  // normalize + store to vals [B][S][H*64] f16
  float inv[8];
#pragma unroll
  for (int r = 0; r < 8; ++r) inv[r] = (l_st[r] > 0.0f) ? 1.0f / l_st[r] : 0.0f;
#pragma unroll
  for (int t = 0; t < 4; ++t)
#pragma unroll
    for (int r = 0; r < 8; ++r) {
      int qg = q0 + wave * 16 + r + 8 * hi;
      int d = t * 16 + n;
      vals[((size_t)b * kS + qg) * kD + h * kHD + d] =
          (_Float16)(accd[t][r] * inv[r]);
    }
}

// ---------------------------------------------------------------------------
// Kernel 3: output projection + ReLU, fp32 out.
// ---------------------------------------------------------------------------
__global__ __launch_bounds__(256) void out_gemm_kernel(const _Float16* __restrict__ vh,
                                                       const _Float16* __restrict__ wh,
                                                       float* __restrict__ out) {
  __shared__ alignas(16) _Float16 As[TM * LDSK];
  __shared__ alignas(16) _Float16 Bs[TN * LDSK];
  const int m0 = blockIdx.x * TM, n0 = blockIdx.y * TN;
  v8f acc[2][4] = {};
  gemm_tile_128x128(vh, wh, m0, n0, As, Bs, acc);

  const int tid = threadIdx.x, wave = tid >> 5, lane = tid & 31;
  const int wm = wave >> 1, wn = wave & 1;
  const int n = lane & 15, hi = lane >> 4;
#pragma unroll
  for (int tm = 0; tm < 2; ++tm)
#pragma unroll
    for (int tn = 0; tn < 4; ++tn) {
      int e = n0 + wn * 64 + tn * 16 + n;
#pragma unroll
      for (int r = 0; r < 8; ++r) {
        int mrow = m0 + wm * 32 + tm * 16 + r + 8 * hi;
        out[(size_t)mrow * kD + e] = fmaxf(acc[tm][tn][r], 0.0f);
      }
    }
}

// ---------------------------------------------------------------------------
// Launcher. Inputs: x [4,2048,1024] f32, mask [4,2048] i32,
// W_qkv [3072,1024] f32, W_comb [1024,1024] f32. Output [4,2048,1024] f32.
// ---------------------------------------------------------------------------
extern "C" void kernel_launch(void* const* d_in, const int* in_sizes, int n_in,
                              void* d_out, int out_size, void* d_ws, size_t ws_size,
                              hipStream_t stream) {
  const float* x     = (const float*)d_in[0];
  const int*   mask  = (const int*)d_in[1];
  const float* Wqkv  = (const float*)d_in[2];
  const float* Wcomb = (const float*)d_in[3];
  float* out = (float*)d_out;

  char* ws = (char*)d_ws;
  size_t off = 0;
  auto take = [&](size_t elems) -> _Float16* {
    _Float16* p = (_Float16*)(ws + off);
    off += (elems * sizeof(_Float16) + 255) & ~(size_t)255;
    return p;
  };
  _Float16* x_h     = take((size_t)kM * kD);                    // 16 MB
  _Float16* wqkv_h  = take((size_t)kNQKV * kD);                 //  6 MB
  _Float16* wcomb_h = take((size_t)kD * kD);                    //  2 MB
  _Float16* qkv_h   = take((size_t)3 * kB * kH * kS * kHD);     // 50 MB
  _Float16* vals_h  = take((size_t)kM * kD);                    // 16 MB
  (void)in_sizes; (void)n_in; (void)out_size; (void)ws_size;

  f32_to_f16_kernel<<<2048, 256, 0, stream>>>(x, x_h, kM * kD / 4);
  f32_to_f16_kernel<<<2048, 256, 0, stream>>>(Wqkv, wqkv_h, kNQKV * kD / 4);
  f32_to_f16_kernel<<<1024, 256, 0, stream>>>(Wcomb, wcomb_h, kD * kD / 4);

  qkv_gemm_kernel<<<dim3(kM / TM, kNQKV / TN), 256, 0, stream>>>(x_h, wqkv_h, qkv_h);
  flash_attn_kernel<<<dim3(kS / 128, kH, kB), 256, 0, stream>>>(qkv_h, mask, vals_h);
  out_gemm_kernel<<<dim3(kM / TM, kD / TN), 256, 0, stream>>>(vals_h, wcomb_h, out);
}
